// MILPAttention_11931419149130
// MI455X (gfx1250) — matrix operand order repaired and checked by reading, not compiled, not run on hardware
//
#include <hip/hip_runtime.h>
#include <math.h>
#include <stdint.h>

#define NTOK   4096
#define DMODEL 1024
#define NHEAD  16
#define DHEAD  64
#define GAMMAF 1.0f

typedef _Float16 half_t;
typedef __attribute__((ext_vector_type(16))) _Float16 v16h;
typedef __attribute__((ext_vector_type(8)))  _Float16 v8h;
typedef __attribute__((ext_vector_type(4)))  _Float16 v4h;
typedef __attribute__((ext_vector_type(8)))  float    v8f;

// ---- WMMA helpers -----------------------------------------------------------

__device__ __forceinline__ v8f wmma_f16(v16h a, v16h b, v8f c) {
  // D = A(16x32 f16) * B(32x16 f16) + C(16x16 f32)
  return __builtin_amdgcn_wmma_f32_16x16x32_f16(false, a, false, b,
                                                (short)0, c, false, false);
}

// A operand: lane holds row (L&15); K elements 16h..16h+15 contiguous
__device__ __forceinline__ v16h load_a_f16(const half_t* __restrict__ row, int koffA) {
  const v8h lo = *(const v8h*)(row + 2 * koffA);
  const v8h hi = *(const v8h*)(row + 2 * koffA + 8);
  v16h r;
#pragma unroll
  for (int i = 0; i < 8; ++i) { r[i] = lo[i]; r[i + 8] = hi[i]; }
  return r;
}

// B operand: lane holds column (L&15); K elements 16h..16h+15 contiguous
__device__ __forceinline__ v16h load_b_f16(const half_t* __restrict__ col, int koffB) {
  return *(const v16h*)(col + koffB);
}

__device__ __forceinline__ v8h shflx16_v8h(v8h v) {
  union { v8h h; int i[4]; } u; u.h = v;
#pragma unroll
  for (int k = 0; k < 4; ++k) u.i[k] = __shfl_xor(u.i[k], 16, 32);
  return u.h;
}

__device__ __forceinline__ v8f vzero8() {
  v8f z;
#pragma unroll
  for (int i = 0; i < 8; ++i) z[i] = 0.0f;
  return z;
}

// ---- f32 -> f16 conversion --------------------------------------------------

__global__ __launch_bounds__(256)
void cvt_kernel(const float* __restrict__ in, half_t* __restrict__ out, int n4) {
  int i = blockIdx.x * blockDim.x + threadIdx.x;
  const int stride = gridDim.x * blockDim.x;
  for (; i < n4; i += stride) {
    const float4 f = ((const float4*)in)[i];
    v4h h;
    h[0] = (half_t)f.x; h[1] = (half_t)f.y; h[2] = (half_t)f.z; h[3] = (half_t)f.w;
    *(v4h*)(out + 4 * (size_t)i) = h;
  }
}

// ---- Fused QKV projection: Q=(xWq^T+bq)/8, K=xWk^T+bk, Vt[h][d][n]=xWv^T+bv --
// Each wave: 32 rows of x (two 16-row tiles) x one head's 64 columns.
// Weight B-tiles amortized over 2 WMMAs each.

__global__ __launch_bounds__(256)
void qkv_kernel(const half_t* __restrict__ xh,
                const half_t* __restrict__ wqh, const half_t* __restrict__ wkh,
                const half_t* __restrict__ wvh,
                const float* __restrict__ bq, const float* __restrict__ bk,
                const float* __restrict__ bv,
                half_t* __restrict__ Qh, half_t* __restrict__ Kh,
                half_t* __restrict__ Vt) {
  const int lane  = threadIdx.x & 31;
  const int wid   = blockIdx.x * (blockDim.x >> 5) + (threadIdx.x >> 5);
  const int head  = wid & (NHEAD - 1);
  const int ntile = wid >> 4;                // 0..127 (32-row tiles)
  const int nbase = ntile * 32;
  const int jbase = head * DHEAD;
  const int l16   = lane & 15;
  const int half_ = lane >> 4;
  const int koffA = half_ * 8;
  const int koffB = half_ * 16;

  v8f accQ[2][4], accK[2][4], accV[2][4];
#pragma unroll
  for (int r = 0; r < 2; ++r)
#pragma unroll
    for (int t = 0; t < 4; ++t) {
      accQ[r][t] = vzero8(); accK[r][t] = vzero8(); accV[r][t] = vzero8();
    }

  const half_t* xrow0 = xh + (size_t)(nbase + l16) * DMODEL;
  const half_t* xrow1 = xrow0 + (size_t)16 * DMODEL;
  const half_t* wq0   = wqh + (size_t)(jbase + l16) * DMODEL;
  const half_t* wk0   = wkh + (size_t)(jbase + l16) * DMODEL;
  const half_t* wv0   = wvh + (size_t)(jbase + l16) * DMODEL;

  for (int kk = 0; kk < DMODEL; kk += 32) {
    __builtin_prefetch(xrow0 + kk + 128, 0, 1);
    const v16h a0 = load_a_f16(xrow0 + kk, koffA);
    const v16h a1 = load_a_f16(xrow1 + kk, koffA);
#pragma unroll
    for (int t = 0; t < 4; ++t) {
      const size_t ro = (size_t)(t * 16) * DMODEL + kk;
      const v16h bqv = load_b_f16(wq0 + ro, koffB);
      accQ[0][t] = wmma_f16(a0, bqv, accQ[0][t]);
      accQ[1][t] = wmma_f16(a1, bqv, accQ[1][t]);
      const v16h bkv = load_b_f16(wk0 + ro, koffB);
      accK[0][t] = wmma_f16(a0, bkv, accK[0][t]);
      accK[1][t] = wmma_f16(a1, bkv, accK[1][t]);
      const v16h bvv = load_b_f16(wv0 + ro, koffB);
      accV[0][t] = wmma_f16(a0, bvv, accV[0][t]);
      accV[1][t] = wmma_f16(a1, bvv, accV[1][t]);
    }
  }

#pragma unroll
  for (int t = 0; t < 4; ++t) {
    const int j = jbase + t * 16 + l16;
    const float bqj = bq[j], bkj = bk[j], bvj = bv[j];
    const int d = t * 16 + l16;
#pragma unroll
    for (int r = 0; r < 2; ++r) {
      const int rb = nbase + r * 16;
#pragma unroll
      for (int v = 0; v < 8; ++v) {
        const int row = rb + v + 8 * half_;
        Qh[(size_t)row * DMODEL + j] = (half_t)((accQ[r][t][v] + bqj) * 0.125f);
        Kh[(size_t)row * DMODEL + j] = (half_t)(accK[r][t][v] + bkj);
      }
      v8h pv;
#pragma unroll
      for (int v = 0; v < 8; ++v) pv[v] = (half_t)(accV[r][t][v] + bvj);
      *(v8h*)(Vt + ((size_t)head * DHEAD + d) * NTOK + rb + 8 * half_) = pv;
    }
  }
}

// ---- Flash attention: O^T[d][n] = sum_m Vt[d][m] * softmax_m(K Q^T - g h)' --
// Each wave: 32 queries (two 16-col tiles). K/Vt A-tiles reused 2x.

__global__ __launch_bounds__(256)
void attn_kernel(const half_t* __restrict__ Qh, const half_t* __restrict__ Kh,
                 const half_t* __restrict__ Vt, const float* __restrict__ hbias,
                 half_t* __restrict__ atth) {
  __shared__ float hb[NTOK];   // raw copy of h; gamma applied at use (fma)
  {
    // Async DMA: 16 KB global -> LDS, 256 threads x 4 x b128 (GV mode).
    const int t = threadIdx.x;
#pragma unroll
    for (int it = 0; it < 4; ++it) {
      const int idx = (t + it * 256) * 4;                 // float index
      unsigned ldsoff = (unsigned)(uintptr_t)(void*)(hb + idx);
      uint64_t gaddr  = (uint64_t)(const void*)(hbias + idx);
      asm volatile("global_load_async_to_lds_b128 %0, %1, off"
                   :: "v"(ldsoff), "v"(gaddr) : "memory");
    }
    asm volatile("s_wait_asynccnt 0" ::: "memory");
  }
  __syncthreads();

  const int lane  = threadIdx.x & 31;
  const int wid   = blockIdx.x * (blockDim.x >> 5) + (threadIdx.x >> 5);
  const int ntile = wid & 127;               // 0..127 (32-query tiles)
  const int head  = wid >> 7;
  const int nbase = ntile * 32;
  const int hoff  = head * DHEAD;
  const int l16   = lane & 15;
  const int half_ = lane >> 4;
  const int koffA = half_ * 8;
  const int koffB = half_ * 16;

  // Q tiles as B operands: [nt][kstep]
  v16h qb[2][2];
#pragma unroll
  for (int nt = 0; nt < 2; ++nt) {
    const half_t* qrow = Qh + (size_t)(nbase + nt * 16 + l16) * DMODEL + hoff;
    qb[nt][0] = load_b_f16(qrow, koffB);
    qb[nt][1] = load_b_f16(qrow + 32, koffB);
  }

  v8f acc[2][4];
#pragma unroll
  for (int nt = 0; nt < 2; ++nt)
#pragma unroll
    for (int t = 0; t < 4; ++t) acc[nt][t] = vzero8();
  float mrun[2] = {-INFINITY, -INFINITY};
  float lrun[2] = {0.0f, 0.0f};

  const half_t* vbase = Vt + (size_t)head * DHEAD * NTOK;

  for (int mb = 0; mb < NTOK; mb += 32) {
    // S^T tiles [mt][nt]: rows = keys, cols = queries (per-lane stats)
    v8f s[2][2];
#pragma unroll
    for (int mt = 0; mt < 2; ++mt) {
      const half_t* krow = Kh + (size_t)(mb + mt * 16 + l16) * DMODEL + hoff;
      __builtin_prefetch(krow + 32 * DMODEL, 0, 1);
      const v16h aLo = load_a_f16(krow, koffA);
      const v16h aHi = load_a_f16(krow + 32, koffA);
#pragma unroll
      for (int nt = 0; nt < 2; ++nt) {
        v8f sv = vzero8();
        sv = wmma_f16(aLo, qb[nt][0], sv);
        sv = wmma_f16(aHi, qb[nt][1], sv);
        s[mt][nt] = sv;
      }
    }
    // bias values (shared across both query tiles)
    const int mrow = mb + 8 * half_;
    float bias0[8], bias1[8];
#pragma unroll
    for (int v = 0; v < 8; ++v) {
      bias0[v] = GAMMAF * hb[mrow + v];
      bias1[v] = GAMMAF * hb[mrow + 16 + v];
    }

    v16h pb[2];
#pragma unroll
    for (int nt = 0; nt < 2; ++nt) {
      float mx = -INFINITY;
#pragma unroll
      for (int v = 0; v < 8; ++v) {
        s[0][nt][v] -= bias0[v];
        s[1][nt][v] -= bias1[v];
        mx = fmaxf(mx, fmaxf(s[0][nt][v], s[1][nt][v]));
      }
      mx = fmaxf(mx, __shfl_xor(mx, 16, 32));
      const float mnew = fmaxf(mrun[nt], mx);
      const float fac  = __expf(mrun[nt] - mnew);

      float lsum = 0.0f;
      v8h h0, h1;
#pragma unroll
      for (int v = 0; v < 8; ++v) {
        const float p0 = __expf(s[0][nt][v] - mnew);
        const float p1 = __expf(s[1][nt][v] - mnew);
        lsum += p0 + p1;
        h0[v] = (half_t)p0; h1[v] = (half_t)p1;
      }
      lsum += __shfl_xor(lsum, 16, 32);
      lrun[nt] = lrun[nt] * fac + lsum;
      mrun[nt] = mnew;
#pragma unroll
      for (int t = 0; t < 4; ++t)
#pragma unroll
        for (int v = 0; v < 8; ++v) acc[nt][t][v] *= fac;

      // P as B operand: lanes 0-15 = keys 0..15 (tile0 own+partner),
      // lanes 16-31 = keys 16..31 (tile1 partner+own); one xor-16 exchange.
      const v8h r0 = shflx16_v8h(h0);
      const v8h r1 = shflx16_v8h(h1);
#pragma unroll
      for (int v = 0; v < 8; ++v) {
        pb[nt][v]     = half_ ? r1[v] : h0[v];
        pb[nt][v + 8] = half_ ? h1[v] : r0[v];
      }
    }
    // O^T += Vt_tile(16d x 32m) * P(32m x 16n), Vt tile reused for both nt
#pragma unroll
    for (int t = 0; t < 4; ++t) {
      const half_t* vrow = vbase + (size_t)(t * 16 + l16) * NTOK + mb;
      const v16h av = load_a_f16(vrow, koffA);
      acc[0][t] = wmma_f16(av, pb[0], acc[0][t]);
      acc[1][t] = wmma_f16(av, pb[1], acc[1][t]);
    }
  }

#pragma unroll
  for (int nt = 0; nt < 2; ++nt) {
    const float inv = 1.0f / lrun[nt];
    half_t* obase = atth + (size_t)(nbase + nt * 16 + l16) * DMODEL + hoff;
#pragma unroll
    for (int t = 0; t < 4; ++t) {
      v8h o;
#pragma unroll
      for (int v = 0; v < 8; ++v) o[v] = (half_t)(acc[nt][t][v] * inv);
      *(v8h*)(obase + t * 16 + 8 * half_) = o;  // slots are consecutive d
    }
  }
}

// ---- Output projection: out = att Wo^T + bo + x (f32) -----------------------
// Each wave: 32 rows x one 64-column strip; Wo tiles amortized 2x.

__global__ __launch_bounds__(256)
void out_kernel(const half_t* __restrict__ atth, const half_t* __restrict__ woh,
                const float* __restrict__ bo, const float* __restrict__ x,
                float* __restrict__ out) {
  const int lane  = threadIdx.x & 31;
  const int wid   = blockIdx.x * (blockDim.x >> 5) + (threadIdx.x >> 5);
  const int strip = wid & 15;
  const int ntile = wid >> 4;                // 0..127
  const int nbase = ntile * 32;
  const int jbase = strip * 64;
  const int l16   = lane & 15;
  const int half_ = lane >> 4;
  const int koffA = half_ * 8;
  const int koffB = half_ * 16;

  v8f acc[2][4];
#pragma unroll
  for (int r = 0; r < 2; ++r)
#pragma unroll
    for (int t = 0; t < 4; ++t) acc[r][t] = vzero8();

  const half_t* arow0 = atth + (size_t)(nbase + l16) * DMODEL;
  const half_t* arow1 = arow0 + (size_t)16 * DMODEL;
  const half_t* wo0   = woh + (size_t)(jbase + l16) * DMODEL;

  for (int kk = 0; kk < DMODEL; kk += 32) {
    __builtin_prefetch(arow0 + kk + 128, 0, 1);
    const v16h a0 = load_a_f16(arow0 + kk, koffA);
    const v16h a1 = load_a_f16(arow1 + kk, koffA);
#pragma unroll
    for (int t = 0; t < 4; ++t) {
      const v16h b = load_b_f16(wo0 + (size_t)(t * 16) * DMODEL + kk, koffB);
      acc[0][t] = wmma_f16(a0, b, acc[0][t]);
      acc[1][t] = wmma_f16(a1, b, acc[1][t]);
    }
  }

#pragma unroll
  for (int t = 0; t < 4; ++t) {
    const int j = jbase + t * 16 + l16;
    const float boj = bo[j];
#pragma unroll
    for (int r = 0; r < 2; ++r) {
#pragma unroll
      for (int v = 0; v < 8; ++v) {
        const int row = nbase + r * 16 + v + 8 * half_;
        const size_t idx = (size_t)row * DMODEL + j;
        out[idx] = acc[r][t][v] + boj + x[idx];
      }
    }
  }
}

// ---- Host-side launch -------------------------------------------------------

extern "C" void kernel_launch(void* const* d_in, const int* in_sizes, int n_in,
                              void* d_out, int out_size, void* d_ws, size_t ws_size,
                              hipStream_t stream) {
  (void)in_sizes; (void)n_in; (void)out_size; (void)ws_size;
  const float* x  = (const float*)d_in[0];
  const float* h  = (const float*)d_in[1];
  const float* Wq = (const float*)d_in[2];
  const float* bq = (const float*)d_in[3];
  const float* Wk = (const float*)d_in[4];
  const float* bk = (const float*)d_in[5];
  const float* Wv = (const float*)d_in[6];
  const float* bv = (const float*)d_in[7];
  const float* Wo = (const float*)d_in[8];
  const float* bo = (const float*)d_in[9];
  float* out = (float*)d_out;

  char* ws = (char*)d_ws;
  size_t off = 0;
  auto carve = [&](size_t bytes) {
    char* p = ws + off;
    off += (bytes + 255) & ~(size_t)255;
    return p;
  };
  half_t* xh   = (half_t*)carve((size_t)NTOK * DMODEL * 2);
  half_t* wqh  = (half_t*)carve((size_t)DMODEL * DMODEL * 2);
  half_t* wkh  = (half_t*)carve((size_t)DMODEL * DMODEL * 2);
  half_t* wvh  = (half_t*)carve((size_t)DMODEL * DMODEL * 2);
  half_t* woh  = (half_t*)carve((size_t)DMODEL * DMODEL * 2);
  half_t* Qh   = (half_t*)carve((size_t)NTOK * DMODEL * 2);
  half_t* Kh   = (half_t*)carve((size_t)NTOK * DMODEL * 2);
  half_t* Vt   = (half_t*)carve((size_t)NTOK * DMODEL * 2);
  half_t* atth = (half_t*)carve((size_t)NTOK * DMODEL * 2);

  cvt_kernel<<<1024, 256, 0, stream>>>(x, xh, NTOK * DMODEL / 4);
  cvt_kernel<<<512, 256, 0, stream>>>(Wq, wqh, DMODEL * DMODEL / 4);
  cvt_kernel<<<512, 256, 0, stream>>>(Wk, wkh, DMODEL * DMODEL / 4);
  cvt_kernel<<<512, 256, 0, stream>>>(Wv, wvh, DMODEL * DMODEL / 4);
  cvt_kernel<<<512, 256, 0, stream>>>(Wo, woh, DMODEL * DMODEL / 4);

  qkv_kernel<<<256, 256, 0, stream>>>(xh, wqh, wkh, wvh, bq, bk, bv, Qh, Kh, Vt);
  attn_kernel<<<256, 256, 0, stream>>>(Qh, Kh, Vt, h, atth);
  out_kernel<<<256, 256, 0, stream>>>(atth, woh, bo, x, out);
}
